// SSM_12086037971230
// MI455X (gfx1250) — compile-verified, matrix operand order checked by reading, not run on hardware
//
#include <hip/hip_runtime.h>
#include <hip/hip_bf16.h>
#include <math.h>

typedef __attribute__((ext_vector_type(16))) __bf16 v16bf;
typedef __attribute__((ext_vector_type(8)))  float  v8f;
typedef unsigned int u32x4 __attribute__((ext_vector_type(4)));
typedef int          i32x4 __attribute__((ext_vector_type(4)));
typedef int          i32x8 __attribute__((ext_vector_type(8)));

#define LENGTH 8192
#define CH     256
#define SD     16
#define NCOL   288          // 256 (dt) + 16 (B) + 16 (C)
#define NTN    (NCOL/16)    // 18 column tiles
#define KCH    (CH/32)      // 8 k-chunks of 32
#define NB     64           // scan blocks
#define BL     (LENGTH/NB)  // 128 steps per block

// ---- workspace layout (bytes) ----
constexpr size_t OFF_XH  = 0;
constexpr size_t SZ_XH   = (size_t)LENGTH * CH * 2;            // bf16 x
constexpr size_t OFF_WP  = OFF_XH + SZ_XH;
constexpr size_t SZ_WP   = (size_t)NTN * KCH * 32 * 16 * 2;    // bf16 W, fragment order
constexpr size_t OFF_DT  = (OFF_WP + SZ_WP + 255) & ~(size_t)255;
constexpr size_t SZ_DT   = (size_t)LENGTH * CH * 4;            // f32 dt
constexpr size_t OFF_BV  = OFF_DT + SZ_DT;
constexpr size_t SZ_BV   = (size_t)LENGTH * SD * 4;
constexpr size_t OFF_CV  = OFF_BV + SZ_BV;
constexpr size_t SZ_CV   = (size_t)LENGTH * SD * 4;
constexpr size_t OFF_AGG = OFF_CV + SZ_CV;
constexpr size_t SZ_AGG  = (size_t)NB * CH * SD * 2 * 4;       // (prodA, h) per block
constexpr size_t OFF_CAR = OFF_AGG + SZ_AGG;

// -------- prep: bf16 convert x; pack W into per-lane WMMA B-fragment order --------
// Wp element index: (((tn*KCH + kc)*32 + lane)*16 + j)  -> B[k(lane,kc,j)][tn*16 + (lane&15)]
__global__ void pack_kernel(const float* __restrict__ x,
                            const float* __restrict__ Wdt,
                            const float* __restrict__ Wb,
                            const float* __restrict__ Wc,
                            __bf16* __restrict__ xh, __bf16* __restrict__ Wp) {
  int i = blockIdx.x * blockDim.x + threadIdx.x;
  if (i < LENGTH * CH) xh[i] = (__bf16)x[i];
  if (i < NTN * KCH * 32 * 16) {
    int j    = i & 15;
    int lane = (i >> 4) & 31;
    int kc   = (i >> 9) & (KCH - 1);
    int tn   = i >> 12;
    int hl = lane >> 4, lr = lane & 15;
    int r = j >> 1, p = j & 1;
    int k = kc * 32 + ((r < 4) ? (2 * r) : (16 + 2 * (r - 4))) + hl * 8 + p;
    int n = tn * 16 + lr;
    float w;
    if (n < CH)            w = Wdt[k * CH + n];
    else if (n < CH + SD)  w = Wb[k * SD + (n - CH)];
    else                   w = Wc[k * SD + (n - CH - SD)];
    Wp[i] = (__bf16)w;
  }
}

// -------- fused GEMM: TDM x-tile -> LDS, WMMA bf16 (dual accumulators), epilogue --------
// One wave per 16-row block; loops over all 18 column tiles sharing the LDS x tile.
__global__ void __launch_bounds__(32)
gemm_wmma_kernel(const __bf16* __restrict__ xh, const __bf16* __restrict__ Wp,
                 const float* __restrict__ b_dt, const float* __restrict__ b_B,
                 const float* __restrict__ b_C,
                 float* __restrict__ dt, float* __restrict__ Bv, float* __restrict__ Cv) {
  __shared__ __bf16 tileA[16 * CH];   // 8 KB; at LDS offset 0

  const int tm   = blockIdx.x;
  const int lane = threadIdx.x;
  const int hl   = lane >> 4;    // 0: lanes 0-15, 1: lanes 16-31
  const int lr   = lane & 15;

#if defined(__HIP_DEVICE_COMPILE__) && __has_builtin(__builtin_amdgcn_tensor_load_to_lds)
  {
    // Tensor DMA: 16x256 bf16 tile of x into LDS (D# per CDNA5 ISA 8.3/8.4)
    unsigned long long ga = (unsigned long long)(uintptr_t)(xh + (size_t)tm * 16 * CH);
    u32x4 g0;
    g0[0] = 1u;                                         // count=1, user mode
    g0[1] = 0u;                                         // lds_addr = 0 (tileA at base)
    g0[2] = (unsigned)(ga & 0xFFFFFFFFu);               // global_addr[31:0]
    g0[3] = (unsigned)((ga >> 32) & 0x1FFFFFFu) | (2u << 30);  // addr[56:32] | type=2
    i32x8 g1;
    g1[0] = (1 << 16);                                  // data_size=2B; mask/pad=0
    g1[1] = (int)(((unsigned)CH & 0xFFFFu) << 16);      // tensor_dim0[15:0]=256
    g1[2] = (int)(((unsigned)LENGTH & 0xFFFFu) << 16);  // dim0 hi=0 | tensor_dim1 lo=8192
    g1[3] = (int)(((unsigned)CH) << 16);                // dim1 hi=0 | tile_dim0=256
    g1[4] = 16;                                         // tile_dim1=16, tile_dim2=0
    g1[5] = CH;                                         // tensor_dim0_stride=256
    g1[6] = 0;
    g1[7] = 0;
    i32x4 gz = {0, 0, 0, 0};
#if __clang_major__ >= 23
    i32x8 gz8 = {0, 0, 0, 0, 0, 0, 0, 0};
    __builtin_amdgcn_tensor_load_to_lds(g0, g1, gz, gz, gz8, 0);
#else
    __builtin_amdgcn_tensor_load_to_lds(g0, g1, gz, gz, 0);
#endif
    __builtin_amdgcn_s_wait_tensorcnt(0);
    // Escape tileA into the asm so the compiler must assume the DMA wrote it:
    // a bare memory clobber is not enough for a non-escaping LDS object (loads
    // from it were folded to undef in the previous build).
    asm volatile("" :: "v"((const void*)tileA) : "memory");
  }
#else
  {
    // fallback: cooperative 16-byte copies
    const uint4* src = (const uint4*)(xh + (size_t)tm * 16 * CH);
    uint4* dst = (uint4*)tileA;
    for (int e = lane; e < 16 * CH / 8; e += 32) dst[e] = src[e];
    __syncthreads();
  }
#endif

  for (int tn = 0; tn < NTN; ++tn) {
    // two independent accumulator chains: breaks the WMMA D->C RAW hazard
    // (bf16 WMMA->WMMA needs 5 NOPs per ISA 7.12.1) so consecutive WMMAs co-execute
    v8f acc0 = {}, acc1 = {};
    #pragma unroll
    for (int kc = 0; kc < KCH; kc += 2) {
      union { uint4 q[2]; v16bf v; } ua0, ub0, ua1, ub1;
      // A fragments: two 16B LDS reads each (K mapping per ISA 7.12.2)
      const __bf16* ap0 = tileA + lr * CH + (kc + 0) * 32 + hl * 8;
      const __bf16* ap1 = tileA + lr * CH + (kc + 1) * 32 + hl * 8;
      ua0.q[0] = *(const uint4*)(ap0);
      ua0.q[1] = *(const uint4*)(ap0 + 16);
      ua1.q[0] = *(const uint4*)(ap1);
      ua1.q[1] = *(const uint4*)(ap1 + 16);
      // B fragments: contiguous 32B per lane from pre-packed W
      const __bf16* bp0 = Wp + ((size_t)(tn * KCH + kc + 0) * 32 + lane) * 16;
      const __bf16* bp1 = Wp + ((size_t)(tn * KCH + kc + 1) * 32 + lane) * 16;
      ub0.q[0] = *(const uint4*)(bp0);
      ub0.q[1] = *(const uint4*)(bp0 + 8);
      ub1.q[0] = *(const uint4*)(bp1);
      ub1.q[1] = *(const uint4*)(bp1 + 8);
      acc0 = __builtin_amdgcn_wmma_f32_16x16x32_bf16(false, ua0.v, false, ub0.v,
                                                     (short)0, acc0, false, false);
      acc1 = __builtin_amdgcn_wmma_f32_16x16x32_bf16(false, ua1.v, false, ub1.v,
                                                     (short)0, acc1, false, false);
    }
    // epilogue: C/D layout — lane half 0: M=r, half 1: M=8+r; N = lr
    const int n = tn * 16 + lr;
    #pragma unroll
    for (int r = 0; r < 8; ++r) {
      const int l = tm * 16 + hl * 8 + r;
      const float z = acc0[r] + acc1[r];
      if (n < CH) {
        float t  = 0.01f + z + b_dt[n];                 // 1/SAMPLE_RATE = 0.01
        float sp = (t > 20.f) ? t : log1pf(__expf(t));  // softplus
        dt[(size_t)l * CH + n] = sp;
      } else if (n < CH + SD) {
        Bv[(size_t)l * SD + (n - CH)] = 1.0f + z + b_B[n - CH];
      } else {
        Cv[(size_t)l * SD + (n - CH - SD)] = z + b_C[n - CH - SD];
      }
    }
  }
}

// ---------------- scan phase 1: per-block (prodA, partial h) ----------------
__global__ void scan_phase1(const float* __restrict__ x, const float* __restrict__ logA,
                            const float* __restrict__ dt, const float* __restrict__ Bv,
                            float2* __restrict__ agg) {
  const int t = blockIdx.x * blockDim.x + threadIdx.x;  // t = b*4096 + c*16 + s
  const int s = t & 15;
  const int c = (t >> 4) & (CH - 1);
  const int b = t >> 12;
  const float As   = -__expf(logA[s]);
  const float invA = 1.0f / As;
  float prodA = 1.0f, h = 0.0f;
  const int l0 = b * BL;
  for (int i = 0; i < BL; ++i) {
    const int l = l0 + i;
    if (i + 16 < BL) __builtin_prefetch(&dt[(size_t)(l + 16) * CH + c], 0, 1);
    const float dtv = dt[(size_t)l * CH + c];
    const float ad  = __expf(dtv * As);
    const float bb  = Bv[(size_t)l * SD + s];
    const float u   = __expf((ad - 1.0f) * bb * invA) * x[(size_t)l * CH + c];
    h = ad * h + u;
    prodA *= ad;
  }
  agg[t] = make_float2(prodA, h);
}

// ---------------- scan phase 2: scan block aggregates -> carries ----------------
__global__ void scan_phase2(const float2* __restrict__ agg, float* __restrict__ carry) {
  const int t = blockIdx.x * blockDim.x + threadIdx.x;  // t = c*16 + s, < 4096
  float h = 0.0f;
  for (int b = 0; b < NB; ++b) {
    carry[(size_t)b * CH * SD + t] = h;                 // state entering block b
    const float2 a = agg[(size_t)b * CH * SD + t];
    h = a.x * h + a.y;
  }
}

// ---------------- scan phase 3: replay + fused y = einsum('ls,lcs->lc') ----------------
__global__ void __launch_bounds__(256)
scan_phase3(const float* __restrict__ x, const float* __restrict__ logA,
            const float* __restrict__ dt, const float* __restrict__ Bv,
            const float* __restrict__ Cv, const float* __restrict__ carry,
            float* __restrict__ y) {
  const int s  = threadIdx.x & 15;        // state lane
  const int cl = threadIdx.x >> 4;        // channel within group (0..15)
  const int cg = blockIdx.x % (CH / 16);
  const int b  = blockIdx.x / (CH / 16);
  const int c  = cg * 16 + cl;
  const float As   = -__expf(logA[s]);
  const float invA = 1.0f / As;
  float h = carry[(size_t)b * CH * SD + c * SD + s];
  const int l0 = b * BL;
  for (int i = 0; i < BL; ++i) {
    const int l = l0 + i;
    const float dtv = dt[(size_t)l * CH + c];
    const float ad  = __expf(dtv * As);
    const float bb  = Bv[(size_t)l * SD + s];
    const float u   = __expf((ad - 1.0f) * bb * invA) * x[(size_t)l * CH + c];
    h = ad * h + u;
    float contrib = Cv[(size_t)l * SD + s] * h;
    // reduce over the 16 state lanes (wave32: two channels per wave, xor masks stay in-group)
    contrib += __shfl_xor(contrib, 1, 32);
    contrib += __shfl_xor(contrib, 2, 32);
    contrib += __shfl_xor(contrib, 4, 32);
    contrib += __shfl_xor(contrib, 8, 32);
    if (s == 0) y[(size_t)l * CH + c] = contrib;
  }
}

extern "C" void kernel_launch(void* const* d_in, const int* in_sizes, int n_in,
                              void* d_out, int out_size, void* d_ws, size_t ws_size,
                              hipStream_t stream) {
  const float* x    = (const float*)d_in[0];
  const float* logA = (const float*)d_in[1];
  const float* Wdt  = (const float*)d_in[2];
  const float* bdt  = (const float*)d_in[3];
  const float* Wb   = (const float*)d_in[4];
  const float* bB   = (const float*)d_in[5];
  const float* Wc   = (const float*)d_in[6];
  const float* bC   = (const float*)d_in[7];
  float* y = (float*)d_out;

  char* ws = (char*)d_ws;
  __bf16* xh   = (__bf16*)(ws + OFF_XH);
  __bf16* Wp   = (__bf16*)(ws + OFF_WP);
  float*  dt   = (float*)(ws + OFF_DT);
  float*  Bv   = (float*)(ws + OFF_BV);
  float*  Cv   = (float*)(ws + OFF_CV);
  float2* agg  = (float2*)(ws + OFF_AGG);
  float*  car  = (float*)(ws + OFF_CAR);

  // prep
  pack_kernel<<<(LENGTH * CH) / 256, 256, 0, stream>>>(x, Wdt, Wb, Wc, xh, Wp);
  // fused GEMM: one wave per 16-row block (TDM x-tile into LDS, 18 col tiles each)
  gemm_wmma_kernel<<<LENGTH / 16, 32, 0, stream>>>(xh, Wp, bdt, bB, bC, dt, Bv, Cv);
  // chunked associative scan
  scan_phase1<<<(NB * CH * SD) / 256, 256, 0, stream>>>(x, logA, dt, Bv, agg);
  scan_phase2<<<(CH * SD) / 256, 256, 0, stream>>>(agg, car);
  scan_phase3<<<NB * (CH / 16), 256, 0, stream>>>(x, logA, dt, Bv, Cv, car, y);
}
